// CommunityHOP_12352325943366
// MI455X (gfx1250) — compile-verified, matrix operand order checked by reading, not run on hardware
//
#include <hip/hip_runtime.h>
#include <hip/hip_bf16.h>
#include <math.h>

// ---------------------------------------------------------------------------
// Types for CDNA5 WMMA (wave32): v_wmma_f32_16x16x32_bf16
// ---------------------------------------------------------------------------
typedef __bf16 bf16_t;
typedef bf16_t v16bf __attribute__((ext_vector_type(16)));
typedef float  v8f   __attribute__((ext_vector_type(8)));
typedef unsigned int v4u __attribute__((ext_vector_type(4)));

#define DEV __device__ __forceinline__

DEV unsigned short f2bf(float f) {              // round-to-nearest-even f32 -> bf16
    unsigned u = __float_as_uint(f);
    unsigned r = u + 0x7FFFu + ((u >> 16) & 1u);
    return (unsigned short)(r >> 16);
}

union Frag { v16bf v; v4u q[2]; };

// A/B fragment loader for 16-bit WMMA operands, row-major ushort(bf16) memory.
// Per ISA layout: lane half h covers K runs [kb+8h .. kb+8h+7] and [kb+16+8h .. +7].
DEV void load_frag(Frag& f, const unsigned short* __restrict__ base,
                   int ld, int row, int kb, int half) {
    const unsigned short* p = base + (size_t)row * ld + kb + half * 8;
    f.q[0] = *(const v4u*)p;
    f.q[1] = *(const v4u*)(p + 16);
}

// ---------------------------------------------------------------------------
// GEMM: C[M x 256] = A[M x 256](bf16) * Bt[256 x 256](bf16, pre-transposed)
// One wave computes a 16(M) x 64(N) tile: 4 accumulators, A-frag reused 4x.
// ---------------------------------------------------------------------------
__global__ void gemm_bf16_256(const unsigned short* __restrict__ A,
                              const unsigned short* __restrict__ Bt,
                              float* __restrict__ C, int Mtiles) {
    const int lane = threadIdx.x & 31;
    const int gw   = (blockIdx.x * blockDim.x + threadIdx.x) >> 5;
    const int mt   = gw % Mtiles;
    const int ng   = gw / Mtiles;
    if (ng >= 4) return;
    const int half = lane >> 4;
    const int idx  = lane & 15;
    const int m    = mt * 16 + idx;

    v8f acc[4];
    #pragma unroll
    for (int s = 0; s < 4; ++s)
        #pragma unroll
        for (int r = 0; r < 8; ++r) acc[s][r] = 0.0f;

    #pragma unroll
    for (int kb = 0; kb < 256; kb += 32) {
        Frag a; load_frag(a, A, 256, m, kb, half);
        #pragma unroll
        for (int s = 0; s < 4; ++s) {
            Frag b; load_frag(b, Bt, 256, ng * 64 + s * 16 + idx, kb, half);
            acc[s] = __builtin_amdgcn_wmma_f32_16x16x32_bf16(
                false, a.v, false, b.v, (short)0, acc[s], false, false);
        }
    }
    // C/D layout: VGPR r of lane -> (M = r + 8*half, N = lane&15)
    #pragma unroll
    for (int s = 0; s < 4; ++s) {
        const int col = ng * 64 + s * 16 + idx;
        #pragma unroll
        for (int r = 0; r < 8; ++r) {
            const int row = mt * 16 + r + 8 * half;
            C[(size_t)row * 256 + col] = acc[s][r];
        }
    }
}

// ---------------------------------------------------------------------------
// Classifier GEMM: out[M x 40] = Z[M x 1280](bf16) * WtC[48 x 1280](bf16) + bc
// Output N padded 40 -> 48 (3 WMMA subtiles), stores guarded to col < 40.
// ---------------------------------------------------------------------------
__global__ void gemm_cls(const unsigned short* __restrict__ Z,
                         const unsigned short* __restrict__ Bt,
                         const float* __restrict__ bc,
                         float* __restrict__ out, int Mtiles) {
    const int lane = threadIdx.x & 31;
    const int mt   = (blockIdx.x * blockDim.x + threadIdx.x) >> 5;
    if (mt >= Mtiles) return;
    const int half = lane >> 4;
    const int idx  = lane & 15;
    const int m    = mt * 16 + idx;

    v8f acc[3];
    #pragma unroll
    for (int s = 0; s < 3; ++s)
        #pragma unroll
        for (int r = 0; r < 8; ++r) acc[s][r] = 0.0f;

    for (int kb = 0; kb < 1280; kb += 32) {
        Frag a; load_frag(a, Z, 1280, m, kb, half);
        #pragma unroll
        for (int s = 0; s < 3; ++s) {
            Frag b; load_frag(b, Bt, 1280, s * 16 + idx, kb, half);
            acc[s] = __builtin_amdgcn_wmma_f32_16x16x32_bf16(
                false, a.v, false, b.v, (short)0, acc[s], false, false);
        }
    }
    #pragma unroll
    for (int s = 0; s < 3; ++s) {
        const int col = s * 16 + idx;
        if (col < 40) {
            const float bias = bc[col];
            #pragma unroll
            for (int r = 0; r < 8; ++r) {
                const int row = mt * 16 + r + 8 * half;
                out[(size_t)row * 40 + col] = acc[s][r] + bias;
            }
        }
    }
}

// ---------------------------------------------------------------------------
// Pointwise / graph kernels
// ---------------------------------------------------------------------------
__global__ void f32_to_bf16_k(const float* __restrict__ in,
                              unsigned short* __restrict__ out, size_t n) {
    size_t i = (size_t)blockIdx.x * blockDim.x + threadIdx.x;
    if (i < n) out[i] = f2bf(in[i]);
}

// W [K x Nout] f32 -> Wt [NoutPad x K] bf16 (transpose + pad with zeros)
__global__ void wt_conv_k(const float* __restrict__ W, unsigned short* __restrict__ Wt,
                          int K, int Nout, int NoutPad) {
    size_t i = (size_t)blockIdx.x * blockDim.x + threadIdx.x;
    if (i >= (size_t)NoutPad * K) return;
    int n = (int)(i / K), k = (int)(i % K);
    Wt[i] = (n < Nout) ? f2bf(W[(size_t)k * Nout + n]) : (unsigned short)0;
}

__global__ void softmax_att_k(const float* __restrict__ att, float* __restrict__ mask, int n) {
    if (blockIdx.x == 0 && threadIdx.x == 0) {
        float m = att[0];
        for (int i = 1; i < n; ++i) m = fmaxf(m, att[i]);
        float s = 0.0f;
        for (int i = 0; i < n; ++i) { float e = __expf(att[i] - m); mask[i] = e; s += e; }
        float inv = 1.0f / s;
        for (int i = 0; i < n; ++i) mask[i] *= inv;
    }
}

__global__ void zero_u32_k(unsigned* __restrict__ p, int n) {
    int i = blockIdx.x * blockDim.x + threadIdx.x;
    if (i < n) p[i] = 0u;
}

__global__ void count_deg_k(const int* __restrict__ dst, int E, unsigned* __restrict__ deg) {
    int i = blockIdx.x * blockDim.x + threadIdx.x;
    if (i < E) atomicAdd(&deg[dst[i]], 1u);
}

__global__ void make_dinv_k(const unsigned* __restrict__ deg, float* __restrict__ dinv, int n) {
    int i = blockIdx.x * blockDim.x + threadIdx.x;
    if (i < n) {
        float d = (float)(deg[i] + 1u);      // +1 self loop; always > 0
        dinv[i] = rsqrtf(d);
    }
}

// agg[i][c] = h[i][c] * dinv[i]^2   (self-loop term; also initializes agg)
__global__ void init_self_k(const float* __restrict__ h, const float* __restrict__ dinv,
                            float* __restrict__ agg, size_t total) {
    size_t i = (size_t)blockIdx.x * blockDim.x + threadIdx.x;
    if (i < total) {
        float dv = dinv[i >> 8];
        agg[i] = h[i] * dv * dv;
    }
}

// One wave per edge: gather 1KB row h[src], scale, atomic-add into agg[dst].
__global__ void edge_scatter_k(const int* __restrict__ src, const int* __restrict__ dst,
                               int E, const float* __restrict__ h,
                               const float* __restrict__ dinv, float* __restrict__ agg) {
    const int lane = threadIdx.x & 31;
    const int e    = (blockIdx.x * blockDim.x + threadIdx.x) >> 5;
    if (e >= E) return;
    const int s = src[e], d = dst[e];
    const float norm = dinv[s] * dinv[d];
    const float* hp = h   + (size_t)s * 256 + lane * 8;
    float*       ap = agg + (size_t)d * 256 + lane * 8;
    if (e + 1 < E)   // gfx1250 global_prefetch_b8 on next edge's source row
        __builtin_prefetch(h + (size_t)src[e + 1] * 256 + lane * 8, 0, 1);
    const float4 v0 = *(const float4*)hp;
    const float4 v1 = *(const float4*)(hp + 4);
    unsafeAtomicAdd(ap + 0, v0.x * norm);
    unsafeAtomicAdd(ap + 1, v0.y * norm);
    unsafeAtomicAdd(ap + 2, v0.z * norm);
    unsafeAtomicAdd(ap + 3, v0.w * norm);
    unsafeAtomicAdd(ap + 4, v1.x * norm);
    unsafeAtomicAdd(ap + 5, v1.y * norm);
    unsafeAtomicAdd(ap + 6, v1.z * norm);
    unsafeAtomicAdd(ap + 7, v1.w * norm);
}

// out_bf[row*ldOut + colOff + c] = bf16( relu(in[row*256+c] + bias[c]) * mask )
__global__ void post_k(const float* __restrict__ in, const float* __restrict__ bias,
                       const float* __restrict__ maskbuf, int maskIdx,
                       unsigned short* __restrict__ outBf, int ldOut, int colOff,
                       size_t total) {
    size_t i = (size_t)blockIdx.x * blockDim.x + threadIdx.x;
    if (i >= total) return;
    size_t row = i >> 8;
    int    c   = (int)(i & 255);
    float  mv  = (maskIdx >= 0) ? maskbuf[maskIdx] : 1.0f;
    float  v   = fmaxf(in[i] + bias[c], 0.0f) * mv;
    outBf[row * (size_t)ldOut + colOff + c] = f2bf(v);
}

// log_softmax over rows of 40, one wave per row, in-place on d_out
__global__ void logsoftmax40_k(float* __restrict__ out, int N) {
    const int lane = threadIdx.x & 31;
    const int row  = (blockIdx.x * blockDim.x + threadIdx.x) >> 5;
    if (row >= N) return;
    float* p = out + (size_t)row * 40;
    float v0 = p[lane];
    float v1 = (lane + 32 < 40) ? p[lane + 32] : -INFINITY;
    float m = fmaxf(v0, v1);
    #pragma unroll
    for (int o = 16; o > 0; o >>= 1) m = fmaxf(m, __shfl_xor(m, o, 32));
    float s = __expf(v0 - m) + ((lane + 32 < 40) ? __expf(v1 - m) : 0.0f);
    #pragma unroll
    for (int o = 16; o > 0; o >>= 1) s += __shfl_xor(s, o, 32);
    float lg = m + __logf(s);
    p[lane] = v0 - lg;
    if (lane + 32 < 40) p[lane + 32] = v1 - lg;
}

// ---------------------------------------------------------------------------
// Launcher
// ---------------------------------------------------------------------------
static inline int cdiv(long long a, long long b) { return (int)((a + b - 1) / b); }

extern "C" void kernel_launch(void* const* d_in, const int* in_sizes, int n_in,
                              void* d_out, int out_size, void* d_ws, size_t ws_size,
                              hipStream_t stream) {
    (void)in_sizes; (void)n_in; (void)out_size; (void)ws_size;
    const int N = 50000, C = 256, H = 256, E = 1600000, HOPS = 3;
    const int CAT = H * (HOPS + 2);      // 1280
    const int O = 40, OPAD = 48;
    const int Mtiles = N / 16;           // 3125

    const float* x     = (const float*)d_in[0];
    const int*   ei    = (const int*)  d_in[1];
    const int*   nei   = (const int*)  d_in[2];
    const float* W_mlp = (const float*)d_in[3];
    const float* b_mlp = (const float*)d_in[4];
    const float* We1   = (const float*)d_in[5];
    const float* be1   = (const float*)d_in[6];
    const float* We2   = (const float*)d_in[7];
    const float* be2   = (const float*)d_in[8];
    const float* Wh    = (const float*)d_in[9];
    const float* bh    = (const float*)d_in[10];
    const float* att   = (const float*)d_in[11];
    const float* Wc    = (const float*)d_in[12];
    const float* bc    = (const float*)d_in[13];
    float*       out   = (float*)d_out;

    // ---- workspace carve (256B aligned) ----
    char* w = (char*)d_ws; size_t off = 0;
    auto carve = [&](size_t bytes) -> void* {
        void* p = w + off; off = (off + bytes + 255) & ~(size_t)255; return p;
    };
    unsigned short* x_bf   = (unsigned short*)carve((size_t)N * C * 2);
    unsigned short* z_bf   = (unsigned short*)carve((size_t)N * CAT * 2);
    float*          h_f    = (float*)carve((size_t)N * H * 4);
    float*          agg    = (float*)carve((size_t)N * H * 4);
    unsigned short* e_bf   = (unsigned short*)carve((size_t)N * H * 2);
    float*          dinvM  = (float*)carve((size_t)N * 4);
    float*          dinvH  = (float*)carve((size_t)HOPS * N * 4);
    unsigned*       deg    = (unsigned*)carve((size_t)N * 4);
    float*          maskb  = (float*)carve(64);
    unsigned short* Wt_mlp = (unsigned short*)carve((size_t)H * C * 2);
    unsigned short* Wt_e1  = (unsigned short*)carve((size_t)H * C * 2);
    unsigned short* Wt_e2  = (unsigned short*)carve((size_t)H * H * 2);
    unsigned short* Wt_h   = (unsigned short*)carve((size_t)HOPS * H * C * 2);
    unsigned short* Wt_c   = (unsigned short*)carve((size_t)OPAD * CAT * 2);

    const int TB = 256;
    const size_t NH = (size_t)N * H;

    // ---- input conversions ----
    f32_to_bf16_k<<<cdiv((long long)N * C, TB), TB, 0, stream>>>(x, x_bf, (size_t)N * C);
    wt_conv_k<<<cdiv((long long)H * C, TB), TB, 0, stream>>>(W_mlp, Wt_mlp, C, H, H);
    wt_conv_k<<<cdiv((long long)H * C, TB), TB, 0, stream>>>(We1, Wt_e1, C, H, H);
    wt_conv_k<<<cdiv((long long)H * H, TB), TB, 0, stream>>>(We2, Wt_e2, H, H, H);
    for (int i = 0; i < HOPS; ++i)
        wt_conv_k<<<cdiv((long long)H * C, TB), TB, 0, stream>>>(
            Wh + (size_t)i * C * H, Wt_h + (size_t)i * H * C, C, H, H);
    wt_conv_k<<<cdiv((long long)OPAD * CAT, TB), TB, 0, stream>>>(Wc, Wt_c, CAT, O, OPAD);
    softmax_att_k<<<1, 32, 0, stream>>>(att, maskb, HOPS + 2);

    // ---- degree -> dinv per graph ----
    auto build_dinv = [&](const int* dstp, float* dinvp) {
        zero_u32_k<<<cdiv(N, TB), TB, 0, stream>>>(deg, N);
        count_deg_k<<<cdiv(E, TB), TB, 0, stream>>>(dstp, E, deg);
        make_dinv_k<<<cdiv(N, TB), TB, 0, stream>>>(deg, dinvp, N);
    };
    build_dinv(ei + E, dinvM);
    for (int i = 0; i < HOPS; ++i)
        build_dinv(nei + (size_t)i * 2 * E + E, dinvH + (size_t)i * N);

    const int gemmBlocks = cdiv((long long)Mtiles * 4, 8);   // 8 waves/block
    const int ewBlocks   = cdiv((long long)E, 8);            // 8 edge-waves/block
    const int pwBlocks   = cdiv((long long)NH, TB);

    // ---- branch: MLP (no aggregation), z slice 4 ----
    gemm_bf16_256<<<gemmBlocks, TB, 0, stream>>>(x_bf, Wt_mlp, h_f, Mtiles);
    post_k<<<pwBlocks, TB, 0, stream>>>(h_f, b_mlp, maskb, 0, z_bf, CAT, 4 * H, NH);

    // ---- branch: extra conv (2-layer GCN on main graph), z slice 3 ----
    gemm_bf16_256<<<gemmBlocks, TB, 0, stream>>>(x_bf, Wt_e1, h_f, Mtiles);
    init_self_k<<<pwBlocks, TB, 0, stream>>>(h_f, dinvM, agg, NH);
    edge_scatter_k<<<ewBlocks, TB, 0, stream>>>(ei, ei + E, E, h_f, dinvM, agg);
    post_k<<<pwBlocks, TB, 0, stream>>>(agg, be1, maskb, -1, e_bf, H, 0, NH);

    gemm_bf16_256<<<gemmBlocks, TB, 0, stream>>>(e_bf, Wt_e2, h_f, Mtiles);
    init_self_k<<<pwBlocks, TB, 0, stream>>>(h_f, dinvM, agg, NH);
    edge_scatter_k<<<ewBlocks, TB, 0, stream>>>(ei, ei + E, E, h_f, dinvM, agg);
    post_k<<<pwBlocks, TB, 0, stream>>>(agg, be2, maskb, 1, z_bf, CAT, 3 * H, NH);

    // ---- branches: hop convs, z slices 0..2 (mask index i+1 per reference) ----
    for (int i = 0; i < HOPS; ++i) {
        const int*   srcp  = nei + (size_t)i * 2 * E;
        const int*   dstp  = srcp + E;
        const float* dinvp = dinvH + (size_t)i * N;
        gemm_bf16_256<<<gemmBlocks, TB, 0, stream>>>(
            x_bf, Wt_h + (size_t)i * H * C, h_f, Mtiles);
        init_self_k<<<pwBlocks, TB, 0, stream>>>(h_f, dinvp, agg, NH);
        edge_scatter_k<<<ewBlocks, TB, 0, stream>>>(srcp, dstp, E, h_f, dinvp, agg);
        post_k<<<pwBlocks, TB, 0, stream>>>(agg, bh + (size_t)i * H, maskb, i + 1,
                                            z_bf, CAT, i * H, NH);
    }

    // ---- classifier + log_softmax ----
    gemm_cls<<<cdiv(Mtiles, 8), TB, 0, stream>>>(z_bf, Wt_c, bc, out, Mtiles);
    logsoftmax40_k<<<cdiv((long long)N, 8), TB, 0, stream>>>(out, N);
}
